// KNNAttentionLayer_81844896792935
// MI455X (gfx1250) — matrix operand order repaired and checked by reading, not compiled, not run on hardware
//
#include <hip/hip_runtime.h>
#include <hip/hip_bf16.h>
#include <math.h>

// ---------------------------------------------------------------------------
// Problem constants (from reference): B=2, S=1024, D=1024, H=16, HD=64,
// N=32768 store entries, KNN_K=32, TEMP=0.1
// ---------------------------------------------------------------------------
#define DMODEL 1024
#define NHEAD  16
#define HDIM   64
#define KNN_K  32
#define NSTORE 32768
#define BATCH  2
#define SEQ    1024
#define BS     (BATCH * SEQ)   // 2048 token rows
#define TEMP   0.1f

typedef __bf16 bf16;
typedef __attribute__((ext_vector_type(16))) __bf16 v16bf;
typedef __attribute__((ext_vector_type(8)))  float  v8f;
typedef unsigned int u32x4 __attribute__((ext_vector_type(4)));
typedef int          i32x4 __attribute__((ext_vector_type(4)));
typedef int          i32x8 __attribute__((ext_vector_type(8)));

// ---------------------------------------------------------------------------
// WMMA fragment loaders (CDNA5 16x16x32 bf16 layouts, wave32)
// A (16x32): lane L -> row M = L&15; half g = L>>4:
//   VGPR r<4: K = 8g+2r, 8g+2r+1 ; VGPR r>=4: K = 16+8g+2(r-4)...
// B (32x16): lane L -> col N = L&15; half g: same K order as A
// C/D (16x16 f32): lane L -> N = L&15; VGPR r -> M = r + 8*(L>>4)
// ---------------------------------------------------------------------------
__device__ inline v16bf frag_a(const bf16* base, int stride, int k0) {
  const int lane = threadIdx.x & 31;
  const int m = lane & 15, g = lane >> 4;
  union { v16bf v; unsigned u[8]; } f;
  const bf16* rp = base + m * stride + k0;
#pragma unroll
  for (int r = 0; r < 8; ++r) {
    const int k = ((r < 4) ? 0 : 16) + 8 * g + 2 * (r & 3);
    f.u[r] = *(const unsigned*)(rp + k);
  }
  return f.v;
}

__device__ inline v16bf frag_b(const bf16* base, int stride, int k0) {
  const int lane = threadIdx.x & 31;
  const int n = lane & 15, g = lane >> 4;
  union { v16bf v; unsigned u[8]; } f;
  const bf16* cp = base + n * stride + k0;
#pragma unroll
  for (int r = 0; r < 8; ++r) f.u[r] = *(const unsigned*)(cp + ((r < 4) ? 0 : 16) + 8 * g + 2 * (r & 3));
  return f.v;
}

__device__ inline v8f wmma_bf16(v16bf a, v16bf b, v8f c) {
  return __builtin_amdgcn_wmma_f32_16x16x32_bf16(
      false, a, false, b, (short)0, c, false, false);
}

// ---------------------------------------------------------------------------
// Tensor Data Mover: DMA a [rows x cols] bf16 tile (cols = 32 or 64; row
// stride = rowstride elements) from global into LDS at lds_byte, padding
// +16B after each row so LDS row stride = cols + 8 bf16 elements.
// D# layout per CDNA5 ISA. Tracked by TENSORcnt.
// ---------------------------------------------------------------------------
#define LDT 40  // GEMM LDS row stride (32 data + 8 pad bf16)
#define LDF 72  // flash LDS row stride (64 data + 8 pad bf16)

__device__ inline void tdm_load_tile_bf16(const bf16* gsrc, unsigned lds_byte,
                                          int rows, int cols, int rowstride) {
#if defined(__AMDGCN__)
  const unsigned long long ga = (unsigned long long)(size_t)gsrc;
  const unsigned padcode = (cols == 32) ? 3u : 4u;  // 16 or 32 DWORDs per row
  u32x4 g0;
  g0[0] = 1u;                                    // count = 1 valid descriptor
  g0[1] = lds_byte;                              // lds_addr
  g0[2] = (unsigned)(ga & 0xffffffffull);        // global_addr[31:0]
  g0[3] = (unsigned)((ga >> 32) & 0x01ffffffull) // global_addr[56:32]
          | (2u << 30);                          // type = 2 ("image")
  i32x8 g1;
  g1[0] = (int)((1u << 16)          // data_size = 1 -> 2 bytes
                | (1u << 20)        // pad_enable
                | (padcode << 22)   // pad_interval (one row)
                | (3u << 25));      // pad_amount: 4 DWORDs (16B)
  g1[1] = (int)(((unsigned)cols) << 16);   // tensor_dim0[15:0]
  g1[2] = (int)(((unsigned)rows) << 16);   // tensor_dim0 hi=0 | tensor_dim1 lo
  g1[3] = (int)(((unsigned)cols) << 16);   // tensor_dim1 hi=0 | tile_dim0
  g1[4] = rows;                            // tile_dim1 = rows, tile_dim2 = 0
  g1[5] = rowstride;                       // tensor_dim0_stride[31:0]
  g1[6] = 0;
  g1[7] = 0;
  i32x4 gz = {0, 0, 0, 0};
#if defined(__clang_major__) && (__clang_major__ >= 23)
  i32x8 gz8 = {0, 0, 0, 0, 0, 0, 0, 0};
  __builtin_amdgcn_tensor_load_to_lds(g0, g1, gz, gz, gz8, 0);
#else
  __builtin_amdgcn_tensor_load_to_lds(g0, g1, gz, gz, 0);
#endif
#else
  (void)gsrc; (void)lds_byte; (void)rows; (void)cols; (void)rowstride;
#endif
}

__device__ inline void tdm_wait_le2() {
#if defined(__AMDGCN__)
  __builtin_amdgcn_s_wait_tensorcnt(2);
#endif
}
__device__ inline void tdm_wait_all() {
#if defined(__AMDGCN__)
  __builtin_amdgcn_s_wait_tensorcnt(0);
#endif
}

// ---------------------------------------------------------------------------
// Tiled WMMA GEMM (NT), bf16 in / OutT out: C[M,N] = A[M,K]*B[N,K]^T (+bias)
// Block 128x128x32, 256 thr = 8 waves (2x4), wave tile 64x32.
// TDM double-buffered LDS: wave0 issues next tile's 2 DMAs, waits
// tensorcnt<=2 for the current buffer, barrier, 8 WMMAs/wave.
// ---------------------------------------------------------------------------
template <bool HAS_BIAS, typename OutT>
__global__ __launch_bounds__(256) void gemm_nt_bf16(const bf16* __restrict__ A,
                                                    const bf16* __restrict__ Bm,
                                                    const float* __restrict__ bias,
                                                    OutT* __restrict__ C,
                                                    int M, int N, int K) {
  constexpr int BM = 128, BN = 128, BK = 32;
  __shared__ bf16 As[2][BM * LDT];
  __shared__ bf16 Bs[2][BN * LDT];

  const int tid = threadIdx.x;
  const int wid = tid >> 5;
  const int wm = wid >> 2, wn = wid & 3;
  const int row0 = blockIdx.y * BM;
  const int col0 = blockIdx.x * BN;
  const bool issuer = (wid == 0);

  v8f acc[4][2];
#pragma unroll
  for (int i = 0; i < 4; ++i)
#pragma unroll
    for (int j = 0; j < 2; ++j) acc[i][j] = v8f{};

  const bf16* Abase = A + (size_t)row0 * K;
  const bf16* Bbase = Bm + (size_t)col0 * K;
  const unsigned laA0 = (unsigned)(size_t)&As[0][0];
  const unsigned laA1 = (unsigned)(size_t)&As[1][0];
  const unsigned laB0 = (unsigned)(size_t)&Bs[0][0];
  const unsigned laB1 = (unsigned)(size_t)&Bs[1][0];

  const int nk = K / BK;
  if (issuer) {
    tdm_load_tile_bf16(Abase, laA0, BM, BK, K);
    tdm_load_tile_bf16(Bbase, laB0, BN, BK, K);
  }

  for (int ki = 0; ki < nk; ++ki) {
    const int cur = ki & 1;
    if (issuer) {
      if (ki + 1 < nk) {
        const int koff = (ki + 1) * BK;
        tdm_load_tile_bf16(Abase + koff, cur ? laA0 : laA1, BM, BK, K);
        tdm_load_tile_bf16(Bbase + koff, cur ? laB0 : laB1, BN, BK, K);
        tdm_wait_le2();   // current buffer's 2 DMAs complete
      } else {
        tdm_wait_all();
      }
    }
    __syncthreads();

    const bf16* Ac = As[cur];
    const bf16* Bc = Bs[cur];
    v16bf af[4], bfm[2];
#pragma unroll
    for (int mt = 0; mt < 4; ++mt)
      af[mt] = frag_a(&Ac[(wm * 64 + mt * 16) * LDT], LDT, 0);
#pragma unroll
    for (int nt = 0; nt < 2; ++nt)
      bfm[nt] = frag_b(&Bc[(wn * 32 + nt * 16) * LDT], LDT, 0);
#pragma unroll
    for (int mt = 0; mt < 4; ++mt)
#pragma unroll
      for (int nt = 0; nt < 2; ++nt)
        acc[mt][nt] = wmma_bf16(af[mt], bfm[nt], acc[mt][nt]);
    __syncthreads();
  }

  const int lane = tid & 31;
  const int g = lane >> 4;
#pragma unroll
  for (int mt = 0; mt < 4; ++mt) {
    const int rbase = row0 + wm * 64 + mt * 16 + g * 8;
#pragma unroll
    for (int nt = 0; nt < 2; ++nt) {
      const int cn = col0 + wn * 32 + nt * 16 + (lane & 15);
      float bv = 0.f;
      if constexpr (HAS_BIAS) bv = bias[cn];
#pragma unroll
      for (int r = 0; r < 8; ++r)
        C[(size_t)(rbase + r) * N + cn] = (OutT)(acc[mt][nt][r] + bv);
    }
  }
}

// ---------------------------------------------------------------------------
// Flash attention (causal), per head, bf16 q/k/v in, bf16 ctx out.
// 128 thr (4 waves), 64 query rows per block, 64-key tiles; K tile via TDM,
// V transposed-copied manually; WMMA for Q*K^T and P*V; online softmax.
// ---------------------------------------------------------------------------
__global__ __launch_bounds__(128) void flash_attn(const bf16* __restrict__ q,
                                                  const bf16* __restrict__ k,
                                                  const bf16* __restrict__ v,
                                                  bf16* __restrict__ ctx) {
  constexpr int BR = 64, BC = 64;
  __shared__ bf16 Qs[BR * LDF];
  __shared__ bf16 Ks[BC * LDF];      // [key][dim]  (TDM-filled)
  __shared__ bf16 Vts[HDIM * LDF];   // [dim][key]
  __shared__ bf16 Ps[4][16 * LDF];   // per-wave P tile

  const int tid = threadIdx.x;
  const int w = tid >> 5;
  const int lane = tid & 31;
  const int g = lane >> 4;
  const int bh = blockIdx.y;
  const int b = bh / NHEAD, h = bh % NHEAD;
  const int col0 = h * HDIM;
  const int q0 = blockIdx.x * BR;

  for (int i = tid; i < BR * HDIM; i += 128) {
    int r = i >> 6, c = i & 63;
    // 1/sqrt(64) = 0.125 is a power of two: exact in bf16
    Qs[r * LDF + c] =
        (bf16)((float)q[((size_t)b * SEQ + q0 + r) * DMODEL + col0 + c] * 0.125f);
  }

  float mrow[8], lrow[8];
  v8f oacc[4];
#pragma unroll
  for (int r = 0; r < 8; ++r) { mrow[r] = -INFINITY; lrow[r] = 0.f; }
#pragma unroll
  for (int t = 0; t < 4; ++t) oacc[t] = v8f{};

  const unsigned laK = (unsigned)(size_t)&Ks[0];

  for (int jb = 0; jb <= (int)blockIdx.x; ++jb) {
    __syncthreads();   // previous iteration's LDS reads complete
    if (w == 0)
      tdm_load_tile_bf16(k + ((size_t)b * SEQ + jb * BC) * DMODEL + col0,
                         laK, BC, HDIM, DMODEL);
    for (int i = tid; i < BC * HDIM; i += 128) {
      int r = i >> 6, c = i & 63;
      size_t gi = ((size_t)b * SEQ + jb * BC + r) * DMODEL + col0 + c;
      Vts[c * LDF + r] = v[gi];
      if (jb < (int)blockIdx.x)   // prefetch next V tile (global_prefetch_b8)
        __builtin_prefetch(&v[gi + (size_t)BC * DMODEL], 0, 1);
    }
    if (w == 0) tdm_wait_all();
    __syncthreads();

    v16bf qa0 = frag_a(&Qs[(w * 16) * LDF], LDF, 0);
    v16bf qa1 = frag_a(&Qs[(w * 16) * LDF], LDF, 32);
    v8f sacc[4];
#pragma unroll
    for (int nt = 0; nt < 4; ++nt) {
      v16bf kb0 = frag_b(&Ks[(nt * 16) * LDF], LDF, 0);
      v16bf kb1 = frag_b(&Ks[(nt * 16) * LDF], LDF, 32);
      sacc[nt] = wmma_bf16(qa0, kb0, v8f{});
      sacc[nt] = wmma_bf16(qa1, kb1, sacc[nt]);
    }

#pragma unroll
    for (int r = 0; r < 8; ++r) {
      const int qrow = q0 + w * 16 + g * 8 + r;
      float mx = -INFINITY;
#pragma unroll
      for (int nt = 0; nt < 4; ++nt) {
        const int kcol = jb * BC + nt * 16 + (lane & 15);
        float s = sacc[nt][r];
        if (kcol > qrow) s = -INFINITY;
        sacc[nt][r] = s;
        mx = fmaxf(mx, s);
      }
#pragma unroll
      for (int off = 1; off < 16; off <<= 1)
        mx = fmaxf(mx, __shfl_xor(mx, off, 32));
      const float mnew = fmaxf(mrow[r], mx);
      float sum = 0.f;
#pragma unroll
      for (int nt = 0; nt < 4; ++nt) {
        float p = __expf(sacc[nt][r] - mnew);
        sacc[nt][r] = p;
        sum += p;
      }
#pragma unroll
      for (int off = 1; off < 16; off <<= 1) sum += __shfl_xor(sum, off, 32);
      const float alpha = __expf(mrow[r] - mnew);
      lrow[r] = lrow[r] * alpha + sum;
      mrow[r] = mnew;
#pragma unroll
      for (int t = 0; t < 4; ++t) oacc[t][r] *= alpha;
#pragma unroll
      for (int nt = 0; nt < 4; ++nt)
        Ps[w][(g * 8 + r) * LDF + nt * 16 + (lane & 15)] = (bf16)sacc[nt][r];
    }

    v16bf pa0 = frag_a(&Ps[w][0], LDF, 0);
    v16bf pa1 = frag_a(&Ps[w][0], LDF, 32);
#pragma unroll
    for (int t = 0; t < 4; ++t) {
      v16bf vb0 = frag_b(&Vts[(t * 16) * LDF], LDF, 0);
      v16bf vb1 = frag_b(&Vts[(t * 16) * LDF], LDF, 32);
      oacc[t] = wmma_bf16(pa0, vb0, oacc[t]);
      oacc[t] = wmma_bf16(pa1, vb1, oacc[t]);
    }
  }

#pragma unroll
  for (int t = 0; t < 4; ++t) {
    const int cn = col0 + t * 16 + (lane & 15);
#pragma unroll
    for (int r = 0; r < 8; ++r) {
      const int row = q0 + w * 16 + g * 8 + r;
      ctx[((size_t)b * SEQ + row) * DMODEL + cn] = (bf16)(oacc[t][r] / lrow[r]);
    }
  }
}

// ---------------------------------------------------------------------------
// Elementwise f32 -> bf16 (n multiple of 1024; grid n/1024, block 256, x4)
// ---------------------------------------------------------------------------
__global__ __launch_bounds__(256) void cvt_bf16(const float* __restrict__ in,
                                                bf16* __restrict__ outb, int n) {
  const int i = blockIdx.x * 1024 + threadIdx.x * 4;
  if (i + 3 < n) {
    const float4 f = *reinterpret_cast<const float4*>(in + i);
    bf16 o[4] = {(bf16)f.x, (bf16)f.y, (bf16)f.z, (bf16)f.w};
    *reinterpret_cast<uint2*>(outb + i) = *reinterpret_cast<uint2*>(o);
  }
}

// ---------------------------------------------------------------------------
// Row L2-normalize (f32 in -> bf16 out), scale = 1/max(||x||, 1e-12)
// 256 threads x 4 elements = 1024 = DMODEL
// ---------------------------------------------------------------------------
__global__ __launch_bounds__(256) void rownorm_bf16(const float* __restrict__ in,
                                                    bf16* __restrict__ outb) {
  __shared__ float red[256];
  const int row = blockIdx.x, tid = threadIdx.x;
  const float* src = in + (size_t)row * DMODEL;
  const float4 x = *reinterpret_cast<const float4*>(src + tid * 4);
  red[tid] = x.x * x.x + x.y * x.y + x.z * x.z + x.w * x.w;
  __syncthreads();
  for (int s = 128; s > 0; s >>= 1) {
    if (tid < s) red[tid] += red[tid + s];
    __syncthreads();
  }
  const float sc = 1.f / fmaxf(sqrtf(red[0]), 1e-12f);
  bf16 o[4] = {(bf16)(x.x * sc), (bf16)(x.y * sc), (bf16)(x.z * sc),
               (bf16)(x.w * sc)};
  *reinterpret_cast<uint2*>(outb + (size_t)row * DMODEL + tid * 4) =
      *reinterpret_cast<uint2*>(o);
}

// ---------------------------------------------------------------------------
// Top-32 of 32768 per row; sims row staged in 128KB dynamic LDS (WGP 320KB)
// ---------------------------------------------------------------------------
__global__ __launch_bounds__(256) void topk32(const float* __restrict__ sims,
                                              float* __restrict__ tsim,
                                              int* __restrict__ tidx) {
  extern __shared__ float srow[];  // NSTORE floats
  __shared__ float rv[256];
  __shared__ int ri[256];
  const int row = blockIdx.x, tid = threadIdx.x;
  const float* src = sims + (size_t)row * NSTORE;
  for (int i = tid * 4; i < NSTORE; i += 1024)
    *reinterpret_cast<float4*>(srow + i) =
        *reinterpret_cast<const float4*>(src + i);
  __syncthreads();
  for (int it = 0; it < KNN_K; ++it) {
    float bv = -INFINITY;
    int bi = 0x7fffffff;
    for (int i = tid; i < NSTORE; i += 256) {
      float x = srow[i];
      if (x > bv) { bv = x; bi = i; }
    }
    rv[tid] = bv; ri[tid] = bi;
    __syncthreads();
    for (int s = 128; s > 0; s >>= 1) {
      if (tid < s) {
        float ov = rv[tid + s]; int oi = ri[tid + s];
        if (ov > rv[tid] || (ov == rv[tid] && oi < ri[tid])) {
          rv[tid] = ov; ri[tid] = oi;
        }
      }
      __syncthreads();
    }
    if (tid == 0) {
      tsim[(size_t)row * KNN_K + it] = rv[0];
      tidx[(size_t)row * KNN_K + it] = ri[0];
      srow[ri[0]] = -INFINITY;
    }
    __syncthreads();
  }
}

// ---------------------------------------------------------------------------
// Softmax(top_sims/T) weighted gather of store_vals rows -> bf16 result
// (feeds the Wproj GEMM directly). 256 thr x 4 cols, b128 gathers.
// ---------------------------------------------------------------------------
__global__ __launch_bounds__(256) void knn_gather(const float* __restrict__ tsim,
                                                  const int* __restrict__ tidx,
                                                  const float* __restrict__ vals,
                                                  bf16* __restrict__ outb) {
  __shared__ float w[KNN_K];
  __shared__ int id[KNN_K];
  const int row = blockIdx.x, tid = threadIdx.x;
  if (tid == 0) {
    float s[KNN_K], m = -INFINITY, tot = 0.f;
    for (int i = 0; i < KNN_K; ++i) {
      s[i] = tsim[(size_t)row * KNN_K + i] * (1.0f / TEMP);
      m = fmaxf(m, s[i]);
    }
    for (int i = 0; i < KNN_K; ++i) { s[i] = __expf(s[i] - m); tot += s[i]; }
    const float inv = 1.f / tot;
    for (int i = 0; i < KNN_K; ++i) w[i] = s[i] * inv;
  }
  if (tid < KNN_K) id[tid] = tidx[(size_t)row * KNN_K + tid];
  __syncthreads();
  const int d = tid * 4;
  float ax = 0.f, ay = 0.f, az = 0.f, aw = 0.f;
#pragma unroll
  for (int kk = 0; kk < KNN_K; ++kk) {
    const float4 vv =
        *reinterpret_cast<const float4*>(vals + (size_t)id[kk] * DMODEL + d);
    const float wk = w[kk];
    ax += wk * vv.x; ay += wk * vv.y; az += wk * vv.z; aw += wk * vv.w;
  }
  bf16 o[4] = {(bf16)ax, (bf16)ay, (bf16)az, (bf16)aw};
  *reinterpret_cast<uint2*>(outb + (size_t)row * DMODEL + d) =
      *reinterpret_cast<uint2*>(o);
}

// ---------------------------------------------------------------------------
// gate = sigmoid([attn, knn] . Wg + bg); out = g*attn + (1-g)*knn
// 256 thr x 4 cols, float4 throughout.
// ---------------------------------------------------------------------------
__global__ __launch_bounds__(256) void gate_combine(const float* __restrict__ attn,
                                                    const float* __restrict__ knn,
                                                    const float* __restrict__ Wg,
                                                    const float* __restrict__ bg,
                                                    float* __restrict__ out) {
  __shared__ float red[256];
  const int row = blockIdx.x, tid = threadIdx.x, d = tid * 4;
  const float* a = attn + (size_t)row * DMODEL;
  const float* n = knn + (size_t)row * DMODEL;
  const float4 av = *reinterpret_cast<const float4*>(a + d);
  const float4 nv = *reinterpret_cast<const float4*>(n + d);
  const float4 wa = *reinterpret_cast<const float4*>(Wg + d);
  const float4 wn = *reinterpret_cast<const float4*>(Wg + DMODEL + d);
  red[tid] = av.x * wa.x + av.y * wa.y + av.z * wa.z + av.w * wa.w +
             nv.x * wn.x + nv.y * wn.y + nv.z * wn.z + nv.w * wn.w;
  __syncthreads();
  for (int s = 128; s > 0; s >>= 1) {
    if (tid < s) red[tid] += red[tid + s];
    __syncthreads();
  }
  const float gate = 1.f / (1.f + __expf(-(red[0] + bg[0])));
  float4 o;
  o.x = gate * av.x + (1.f - gate) * nv.x;
  o.y = gate * av.y + (1.f - gate) * nv.y;
  o.z = gate * av.z + (1.f - gate) * nv.z;
  o.w = gate * av.w + (1.f - gate) * nv.w;
  *reinterpret_cast<float4*>(out + (size_t)row * DMODEL + d) = o;
}

// ---------------------------------------------------------------------------
// Orchestration
// ---------------------------------------------------------------------------
extern "C" void kernel_launch(void* const* d_in, const int* in_sizes, int n_in,
                              void* d_out, int out_size, void* d_ws,
                              size_t ws_size, hipStream_t stream) {
  (void)in_sizes; (void)n_in; (void)out_size; (void)ws_size;
  const float* x   = (const float*)d_in[0];
  const float* sk  = (const float*)d_in[1];
  const float* sv  = (const float*)d_in[2];
  const float* Wq  = (const float*)d_in[3];
  const float* Wk  = (const float*)d_in[4];
  const float* Wv  = (const float*)d_in[5];
  const float* Wo  = (const float*)d_in[6];
  const float* Wkk = (const float*)d_in[7];
  const float* Wpr = (const float*)d_in[8];
  const float* bpr = (const float*)d_in[9];
  const float* Wg  = (const float*)d_in[10];
  const float* bg  = (const float*)d_in[11];
  float* out = (float*)d_out;

  char* p = (char*)d_ws;
  auto alloc = [&](size_t bytes) {
    char* r = p;
    p += (bytes + 255) & ~(size_t)255;
    return (void*)r;
  };
  const size_t rowsz = (size_t)BS * DMODEL;   // 2048*1024
  const size_t wsz = (size_t)DMODEL * DMODEL; // 1024*1024
  bf16* qb        = (bf16*)alloc(sizeof(bf16) * rowsz);
  bf16* kb_       = (bf16*)alloc(sizeof(bf16) * rowsz);
  bf16* vb        = (bf16*)alloc(sizeof(bf16) * rowsz);
  bf16* ctxb      = (bf16*)alloc(sizeof(bf16) * rowsz);
  bf16* knnresb   = (bf16*)alloc(sizeof(bf16) * rowsz);
  float* knnq     = (float*)alloc(sizeof(float) * rowsz);
  float* attn_out = (float*)alloc(sizeof(float) * rowsz);
  float* knn_out  = (float*)alloc(sizeof(float) * rowsz);
  bf16* xb        = (bf16*)alloc(sizeof(bf16) * rowsz);
  bf16* Wqb       = (bf16*)alloc(sizeof(bf16) * wsz);
  bf16* Wkb       = (bf16*)alloc(sizeof(bf16) * wsz);
  bf16* Wvb       = (bf16*)alloc(sizeof(bf16) * wsz);
  bf16* Wob       = (bf16*)alloc(sizeof(bf16) * wsz);
  bf16* Wkkb      = (bf16*)alloc(sizeof(bf16) * wsz);
  bf16* Wprb      = (bf16*)alloc(sizeof(bf16) * wsz);
  bf16* qn        = (bf16*)alloc(sizeof(bf16) * rowsz);
  bf16* kn        = (bf16*)alloc(sizeof(bf16) * (size_t)NSTORE * DMODEL);
  float* sims     = (float*)alloc(sizeof(float) * (size_t)BS * NSTORE);
  float* tsim     = (float*)alloc(sizeof(float) * (size_t)BS * KNN_K);
  int* tidx       = (int*)alloc(sizeof(int) * (size_t)BS * KNN_K);

  const dim3 blk(256);
  const dim3 gProj(DMODEL / 128, BS / 128);  // (8, 16)

  // bf16 staging of GEMM operands (makes every GEMM TDM-eligible)
  cvt_bf16<<<(int)(rowsz / 1024), blk, 0, stream>>>(x, xb, (int)rowsz);
  cvt_bf16<<<(int)(wsz / 1024), blk, 0, stream>>>(Wq, Wqb, (int)wsz);
  cvt_bf16<<<(int)(wsz / 1024), blk, 0, stream>>>(Wk, Wkb, (int)wsz);
  cvt_bf16<<<(int)(wsz / 1024), blk, 0, stream>>>(Wv, Wvb, (int)wsz);
  cvt_bf16<<<(int)(wsz / 1024), blk, 0, stream>>>(Wo, Wob, (int)wsz);
  cvt_bf16<<<(int)(wsz / 1024), blk, 0, stream>>>(Wkk, Wkkb, (int)wsz);
  cvt_bf16<<<(int)(wsz / 1024), blk, 0, stream>>>(Wpr, Wprb, (int)wsz);

  // Projections: q, k, v (bf16 out, feed flash directly), knn_q (f32 out)
  gemm_nt_bf16<false, bf16><<<gProj, blk, 0, stream>>>(xb, Wqb, nullptr, qb, BS, DMODEL, DMODEL);
  gemm_nt_bf16<false, bf16><<<gProj, blk, 0, stream>>>(xb, Wkb, nullptr, kb_, BS, DMODEL, DMODEL);
  gemm_nt_bf16<false, bf16><<<gProj, blk, 0, stream>>>(xb, Wvb, nullptr, vb, BS, DMODEL, DMODEL);
  gemm_nt_bf16<false, float><<<gProj, blk, 0, stream>>>(xb, Wkkb, nullptr, knnq, BS, DMODEL, DMODEL);

  // Causal flash attention (bf16 ctx out) + output projection
  flash_attn<<<dim3(SEQ / 64, BATCH * NHEAD), dim3(128), 0, stream>>>(qb, kb_, vb, ctxb);
  gemm_nt_bf16<false, float><<<gProj, blk, 0, stream>>>(ctxb, Wob, nullptr, attn_out, BS, DMODEL, DMODEL);

  // kNN path: normalize to bf16, TDM+WMMA sims GEMM, top-k, gather, proj
  rownorm_bf16<<<BS, 256, 0, stream>>>(knnq, qn);
  rownorm_bf16<<<NSTORE, 256, 0, stream>>>(sk, kn);
  gemm_nt_bf16<false, float><<<dim3(NSTORE / 128, BS / 128), blk, 0, stream>>>(
      qn, kn, nullptr, sims, BS, NSTORE, DMODEL);
  topk32<<<BS, 256, NSTORE * sizeof(float), stream>>>(sims, tsim, tidx);
  knn_gather<<<BS, 256, 0, stream>>>(tsim, tidx, sv, knnresb);
  gemm_nt_bf16<true, float><<<gProj, blk, 0, stream>>>(knnresb, Wprb, bpr, knn_out, BS, DMODEL, DMODEL);

  // Gated combine -> d_out
  gate_combine<<<BS, 256, 0, stream>>>(attn_out, knn_out, Wg, bg, out);
}
